// AdditiveAttention_62130996904636
// MI455X (gfx1250) — compile-verified, matrix operand order checked by reading, not run on hardware
//
#include <hip/hip_runtime.h>

#define B_    64
#define S_    2048
#define ENC_  512
#define DEC_  512
#define ATTN_ 256

typedef __attribute__((ext_vector_type(16))) __bf16 v16bf;
typedef __attribute__((ext_vector_type(8)))  __bf16 v8bf;
typedef __attribute__((ext_vector_type(2)))  __bf16 v2bf;
typedef __attribute__((ext_vector_type(8)))  float  v8f;
typedef __attribute__((ext_vector_type(4)))  int    v4i;

#if __has_builtin(__builtin_amdgcn_global_load_async_to_lds_b128) && \
    __has_builtin(__builtin_amdgcn_s_wait_asynccnt)
#define USE_ASYNC_LDS 1
typedef __attribute__((address_space(1))) v4i g_v4i;   // global int4
typedef __attribute__((address_space(3))) v4i l_v4i;   // LDS int4
#else
#define USE_ASYNC_LDS 0
#endif

// ---------------------------------------------------------------------------
// Pack W_enc [512,256] f32 -> bf16 in WMMA B-fragment order:
//   p = ((kb*16 + nt)*32 + lane)*16 + i
//   k = kb*32 + (i&7) + 8*(lane>=16) + 16*(i>>3),  n = nt*16 + (lane&15)
// One K-step chunk (kb) is a contiguous 16 KB block -> trivial LDS staging.
// ---------------------------------------------------------------------------
__global__ __launch_bounds__(256) void pack_w_enc_kernel(
    const float* __restrict__ W, __bf16* __restrict__ Wp) {
  int p  = blockIdx.x * 256 + threadIdx.x;   // 0 .. 131071
  int i  = p & 15;
  int L  = (p >> 4) & 31;
  int nt = (p >> 9) & 15;
  int kb = p >> 13;
  int hi = L >> 4;
  int k  = kb * 32 + (i & 7) + 8 * hi + 16 * (i >> 3);
  int n  = nt * 16 + (L & 15);
  Wp[p] = (__bf16)W[k * ATTN_ + n];
}

// ---------------------------------------------------------------------------
// dec_proj[b,n] = sum_k dec_state[b,k] * W_dec[k,n]   (tiny: 64x256, K=512)
// ---------------------------------------------------------------------------
__global__ __launch_bounds__(256) void dec_proj_kernel(
    const float* __restrict__ dec, const float* __restrict__ Wd,
    float* __restrict__ dp) {
  int b = blockIdx.x, n = threadIdx.x;
  const float* drow = dec + (long)b * DEC_;
  float acc = 0.f;
  for (int k = 0; k < DEC_; ++k) acc += drow[k] * Wd[k * ATTN_ + n];
  dp[b * ATTN_ + n] = acc;
}

// ---------------------------------------------------------------------------
// Fused scores: tile M=64 x N=256, 256 threads = 8 waves as 4(M) x 2(N).
// Wave (wm,wn): rows [wm*16, wm*16+16), cols [wn*128, wn*128+128) -> 8 WMMAs.
// Double-buffered LDS staging; fragments preloaded to registers; B chunk
// staged with async global->LDS (ASYNCcnt) when the toolchain exposes it.
// ---------------------------------------------------------------------------
__global__ __launch_bounds__(256) void scores_kernel(
    const float* __restrict__ enc, const __bf16* __restrict__ Wp,
    const float* __restrict__ dp, const float* __restrict__ vvec,
    float* __restrict__ scores) {
  __shared__ __align__(16) __bf16 lA[2][64 * 32];      // 2 x  4 KB A tiles
  __shared__ __align__(16) __bf16 lB[2][16 * 32 * 16]; // 2 x 16 KB W chunks
  __shared__ float wpart[8][16];

  const int tid  = threadIdx.x;
  const int wave = tid >> 5;
  const int lane = tid & 31;
  const int l16  = lane & 15;
  const int hi   = lane >> 4;
  const int wm   = wave >> 1;      // 0..3  (M tile)
  const int wn   = wave & 1;       // 0..1  (N half)

  const long row0 = (long)blockIdx.x * 64;       // flat row in [0, B*S)
  const int  b    = (int)(row0 / S_);            // 64-row tile never crosses b
  const float* encBase = enc + row0 * ENC_;

  auto stageA = [&](int kb, int buf) {
#pragma unroll
    for (int it = 0; it < 4; ++it) {
      int e2 = tid + it * 256;                   // float2 index, 0..1023
      int elem = e2 * 2;                         // half index, 0..2047
      int r = elem >> 5, kk = elem & 31;
      float2 f = *(const float2*)&encBase[(long)r * ENC_ + kb * 32 + kk];
      v2bf h; h[0] = (__bf16)f.x; h[1] = (__bf16)f.y;
      *(v2bf*)&lA[buf][elem] = h;                // v_cvt_pk_bf16_f32 + b32 store
    }
  };
  auto stageB = [&](int kb, int buf) {
    const uint4* src = (const uint4*)(Wp + (size_t)kb * 8192);
#if USE_ASYNC_LDS
    char* dstBytes = (char*)lB[buf];
#pragma unroll
    for (int it = 0; it < 4; ++it) {
      int idx = tid + it * 256;
      __builtin_amdgcn_global_load_async_to_lds_b128(
          (g_v4i*)(src + idx),
          (l_v4i*)(dstBytes + (size_t)idx * 16), 0, 0);
    }
#else
    uint4* dst = (uint4*)lB[buf];
#pragma unroll
    for (int it = 0; it < 4; ++it)
      dst[tid + it * 256] = src[tid + it * 256];
#endif
  };
  auto waitStage = [&]() {
#if USE_ASYNC_LDS
    __builtin_amdgcn_s_wait_asynccnt(0);
#endif
  };

  v8f c[8] = {};                                 // 8 N-tiles of 16x16 f32 accum

  stageA(0, 0);
  stageB(0, 0);
  waitStage();
  __syncthreads();

  for (int kb = 0; kb < ENC_ / 32; ++kb) {
    const int cur = kb & 1, nxt = cur ^ 1;

    // --- Preload this K-step's fragments into registers (batched DS loads) ---
    v8bf alo = *(const v8bf*)&lA[cur][(wm * 16 + l16) * 32 + 8 * hi];
    v8bf ahi = *(const v8bf*)&lA[cur][(wm * 16 + l16) * 32 + 16 + 8 * hi];
    v16bf a;
#pragma unroll
    for (int i = 0; i < 8; ++i) { a[i] = alo[i]; a[i + 8] = ahi[i]; }
    v16bf bfr[8];
#pragma unroll
    for (int j = 0; j < 8; ++j)
      bfr[j] = *(const v16bf*)&lB[cur][((wn * 8 + j) * 32 + lane) * 16];

    // --- Stage next K-step into the other buffer (overlaps the WMMAs) ---
    if (kb + 1 < ENC_ / 32) {
      stageA(kb + 1, nxt);
      stageB(kb + 1, nxt);
    }
    if (kb + 2 < ENC_ / 32) {                    // prefetch what the NEXT stage reads
      if (tid < 64)
        __builtin_prefetch(&encBase[(long)tid * ENC_ + (kb + 2) * 32], 0, 1);
      else if (tid < 192)
        __builtin_prefetch((const char*)Wp + (size_t)(kb + 2) * 16384 +
                               (size_t)(tid - 64) * 128, 0, 1);
    }

    // --- 8 WMMAs from registers, no per-WMMA DS waits ---
#pragma unroll
    for (int j = 0; j < 8; ++j)
      c[j] = __builtin_amdgcn_wmma_f32_16x16x32_bf16(
          false, a, false, bfr[j], (short)0, c[j], false, false);

    waitStage();                                 // retire async B staging
    __syncthreads();                             // one barrier per K-step
  }

  // Epilogue: + dec_proj, tanh, * v[n], sum this lane's 8 columns.
  // C layout: c[j][r] = tile[m = wm*16 + r + 8*hi][n = wn*128 + j*16 + l16].
  float sacc[8];
#pragma unroll
  for (int r = 0; r < 8; ++r) sacc[r] = 0.f;
#pragma unroll
  for (int j = 0; j < 8; ++j) {
    int n = wn * 128 + j * 16 + l16;
    float dpn = dp[b * ATTN_ + n];
    float vn  = vvec[n];
#pragma unroll
    for (int r = 0; r < 8; ++r) sacc[r] += tanhf(c[j][r] + dpn) * vn;
  }
  // Deterministic xor-tree reduction across the 16 lanes of each half-wave.
#pragma unroll
  for (int r = 0; r < 8; ++r) {
    float s = sacc[r];
    for (int m = 8; m >= 1; m >>= 1) s += __shfl_xor(s, m, 16);
    sacc[r] = s;
  }
  if (l16 == 0) {
#pragma unroll
    for (int r = 0; r < 8; ++r) wpart[wave][hi * 8 + r] = sacc[r];
  }
  __syncthreads();
  if (tid < 64) {                                // fixed-order combine of wn=0,1
    int mtile = tid >> 4, rr = tid & 15;
    scores[row0 + tid] = wpart[mtile * 2][rr] + wpart[mtile * 2 + 1][rr];
  }
}

// ---------------------------------------------------------------------------
// Row softmax over S=2048, one block per b. Deterministic tree reductions.
// ---------------------------------------------------------------------------
__global__ __launch_bounds__(256) void softmax_kernel(
    const float* __restrict__ scores, float* __restrict__ attn) {
  __shared__ float red[256];
  int b = blockIdx.x, t = threadIdx.x;
  const float* srow = scores + (long)b * S_;
  float local[8];
  float mx = -3.402823466e+38f;
#pragma unroll
  for (int i = 0; i < 8; ++i) { local[i] = srow[t + i * 256]; mx = fmaxf(mx, local[i]); }
  red[t] = mx; __syncthreads();
  for (int off = 128; off >= 1; off >>= 1) {
    if (t < off) red[t] = fmaxf(red[t], red[t + off]);
    __syncthreads();
  }
  mx = red[0]; __syncthreads();
  float sum = 0.f;
#pragma unroll
  for (int i = 0; i < 8; ++i) { local[i] = __expf(local[i] - mx); sum += local[i]; }
  red[t] = sum; __syncthreads();
  for (int off = 128; off >= 1; off >>= 1) {
    if (t < off) red[t] += red[t + off];
    __syncthreads();
  }
  float inv = 1.f / red[0];
#pragma unroll
  for (int i = 0; i < 8; ++i) attn[(long)b * S_ + t + i * 256] = local[i] * inv;
}

// ---------------------------------------------------------------------------
// context partials: part[b,chunk,e] = sum_{s in chunk} attn[b,s] * enc[b,s,e]
// 32 chunks of 64 s-rows per b (2048 blocks) -> bandwidth-saturating,
// no atomics, deterministic.
// ---------------------------------------------------------------------------
__global__ __launch_bounds__(128) void context_partial_kernel(
    const float* __restrict__ enc, const float* __restrict__ attn,
    float* __restrict__ part) {
  int chunk = blockIdx.x;   // 32
  int b     = blockIdx.y;   // 64
  int t     = threadIdx.x;  // 128
  float acc[4] = {0.f, 0.f, 0.f, 0.f};
  const float* w = attn + (long)b * S_ + chunk * 64;
  const float* e = enc + ((long)b * S_ + chunk * 64) * ENC_;
  for (int s = 0; s < 64; ++s) {
    float ws = w[s];
#pragma unroll
    for (int j = 0; j < 4; ++j) acc[j] += ws * e[(long)s * ENC_ + t + j * 128];
  }
  float* p = part + ((long)b * 32 + chunk) * ENC_;
#pragma unroll
  for (int j = 0; j < 4; ++j) p[t + j * 128] = acc[j];
}

__global__ __launch_bounds__(256) void context_reduce_kernel(
    const float* __restrict__ part, float* __restrict__ ctx) {
  int idx = blockIdx.x * 256 + threadIdx.x;   // 32768 = 64*512
  int b = idx >> 9, e = idx & 511;
  const float* p = part + (long)b * 32 * ENC_ + e;
  float s = 0.f;
#pragma unroll
  for (int c = 0; c < 32; ++c) s += p[c * ENC_];
  ctx[idx] = s;
}

// ---------------------------------------------------------------------------
extern "C" void kernel_launch(void* const* d_in, const int* in_sizes, int n_in,
                              void* d_out, int out_size, void* d_ws, size_t ws_size,
                              hipStream_t stream) {
  const float* enc = (const float*)d_in[0];   // [64,2048,512]
  const float* dec = (const float*)d_in[1];   // [64,512]
  const float* We  = (const float*)d_in[2];   // [512,256]
  const float* Wd  = (const float*)d_in[3];   // [512,256]
  const float* v   = (const float*)d_in[4];   // [256]

  float* out  = (float*)d_out;
  float* ctx  = out;                          // [64,512]   (tuple item 0)
  float* attn = out + B_ * ENC_;              // [64,2048]  (tuple item 1)

  char* ws = (char*)d_ws;
  __bf16* Wp     = (__bf16*)ws;                       // 256 KB packed W_enc
  float*  dp     = (float*)(ws + (256u << 10));       //  64 KB dec_proj
  float*  scores = (float*)(ws + (320u << 10));       // 512 KB scores
  float*  part   = (float*)(ws + (832u << 10));       //   4 MB context partials

  pack_w_enc_kernel<<<512, 256, 0, stream>>>(We, Wp);
  dec_proj_kernel<<<B_, ATTN_, 0, stream>>>(dec, Wd, dp);
  scores_kernel<<<(B_ * S_) / 64, 256, 0, stream>>>(enc, Wp, dp, v, scores);
  softmax_kernel<<<B_, 256, 0, stream>>>(scores, attn);
  context_partial_kernel<<<dim3(32, B_), 128, 0, stream>>>(enc, attn, part);
  context_reduce_kernel<<<(B_ * ENC_) / 256, 256, 0, stream>>>(part, ctx);
}
